// Model_25512105739003
// MI455X (gfx1250) — matrix operand order repaired and checked by reading, not compiled, not run on hardware
//
#include <hip/hip_runtime.h>

typedef __attribute__((ext_vector_type(16))) _Float16 v16h;
typedef __attribute__((ext_vector_type(4)))  _Float16 v4h;
typedef __attribute__((ext_vector_type(8)))  float    v8f;

#define STRIDE 768
#define LAM 0.01f
#define PI2 6.28318530717958647692f

// ---------------- packed weight layout (halfs) ----------------
// wid: 0 Wt1(384x96) 1 Wt2(64x96) 2 Wt3(64x96) 3 Wlstm(32x256)
//      4 Wf1(384x112) 5 Wf2(64x112) 6 Mir(96x64) 7 Wfuse(128x16)
#define W_T1   0
#define W_T2   36864
#define W_T3   43008
#define W_LSTM 49152
#define W_F1   57344
#define W_F2   100352
#define W_MIR  107520
#define W_FUSE 113664
#define W_TOTAL 115712
// bias block (f32) right after weights: byte offset W_TOTAL*2
#define B_T1   0
#define B_T2   96
#define B_T3   192
#define B_LSTM 288
#define B_F1   544
#define B_F2   656
#define B_FZ   768
#define B_ZERO 784   // 16 zeros (bias for irfft)
#define B_TOTAL 800

// ---------------- fast activations (v_exp_f32 + v_rcp_f32, branch-free) ----
__device__ __forceinline__ float frcp_(float x) { return __builtin_amdgcn_rcpf(x); }
__device__ __forceinline__ float sig_(float x)  { return frcp_(1.f + __expf(-x)); }
__device__ __forceinline__ float tanh_(float x) { return 1.f - 2.f * frcp_(1.f + __expf(2.f * x)); }

// ---------------- WMMA helpers ----------------
__device__ __forceinline__ v8f wmma_f16(v16h a, v16h b, v8f c) {
  return __builtin_amdgcn_wmma_f32_16x16x32_f16(false, a, false, b, (short)0, c,
                                                false, false);
}

__device__ __forceinline__ void wave_lds_sync() {
  asm volatile("s_wait_dscnt 0x0" ::: "memory");
}

// A fragment (16x32 f16, MxK) from LDS panel, per ISA A layout:
// lane: m = lane&15, half = lane>>4; VGPR v holds K = (v&4)*4 + half*8 + (v&3)*2 (+1)
__device__ __forceinline__ v16h load_a(const _Float16* base, int col0, int kt) {
  const int lane = threadIdx.x & 31;
  const int m = lane & 15, hf = lane >> 4;
  const _Float16* row = base + m * STRIDE + col0 + kt * 32;
  v16h a;
#pragma unroll
  for (int v = 0; v < 8; ++v) {
    const int k0 = ((v & 4) << 2) + (hf << 3) + ((v & 3) << 1);
    a[2 * v]     = row[k0];
    a[2 * v + 1] = row[k0 + 1];
  }
  return a;
}

// B fragment: prepacked contiguously, one 32B v16h per lane per (kt,nt)
__device__ __forceinline__ v16h load_b(const _Float16* w, int kt, int nTiles, int nt) {
  const int lane = threadIdx.x & 31;
  return *(const v16h*)(w + ((((kt * nTiles + nt) * 32) + lane) << 4));
}

// generic GEMM stage: C[m,n] = sum_k A[m,k]*W[k,n] + bias[n], raw f16 out to LDS
template <int KT, int NT>
__device__ __forceinline__ void gemm_stage(const _Float16* B, int acol,
                                           const _Float16* w, const float* biasp,
                                           _Float16* Bo, int ocol) {
  const int lane = threadIdx.x & 31;
  const int n15 = lane & 15, hf = lane >> 4;
  v8f acc[NT];
#pragma unroll
  for (int nt = 0; nt < NT; ++nt) {
    const float bv = biasp[nt * 16 + n15];
#pragma unroll
    for (int i = 0; i < 8; ++i) acc[nt][i] = bv;
  }
#pragma unroll
  for (int kt = 0; kt < KT; ++kt) {
    const v16h a = load_a(B, acol, kt);
#pragma unroll
    for (int nt = 0; nt < NT; ++nt)
      acc[nt] = wmma_f16(a, load_b(w, kt, NT, nt), acc[nt]);
  }
#pragma unroll
  for (int nt = 0; nt < NT; ++nt) {
    const int n = nt * 16 + n15;
#pragma unroll
    for (int v = 0; v < 8; ++v)
      Bo[(v + hf * 8) * STRIDE + ocol + n] = (_Float16)acc[nt][v];
  }
}

// maxpool over position pairs (2j,2j+1) per channel; pad [C*outL..padK) with 0
__device__ __forceinline__ void pool_pass(_Float16* B, int colRaw, int C, int L,
                                          int outL, int colDst, int padK) {
  const int lane = threadIdx.x & 31;
  const int outF = C * outL;
  for (int idx = lane; idx < 16 * padK; idx += 32) {
    const int m = idx / padK, f = idx % padK;
    float val = 0.f;
    if (f < outF) {
      const int o = f / outL, j = f % outL;
      const float x0 = (float)B[m * STRIDE + colRaw + o * L + 2 * j];
      const float x1 = (float)B[m * STRIDE + colRaw + o * L + 2 * j + 1];
      val = fmaxf(x0, x1);
    }
    B[m * STRIDE + colDst + f] = (_Float16)val;
  }
}

// single-step LSTM (h0=c0=0): gates = x@wih.T + (bih+bhh); h = sig(o)*tanh(sig(i)*tanh(g))
// ISFREQ: write softshrunk spectrum v[66] (real[0:33], imag[0:33]) at outCol
template <bool ISFREQ>
__device__ __forceinline__ void lstm_stage(const _Float16* Br, int acol,
                                           const _Float16* w, const float* bias,
                                           _Float16* B, int outCol) {
  const int lane = threadIdx.x & 31;
  const int n15 = lane & 15, hf = lane >> 4;
  const v16h a = load_a(Br, acol, 0);
#pragma unroll
  for (int t = 0; t < 4; ++t) {
    v8f ai, ag, ao;
    const float bi = bias[t * 16 + n15];
    const float bg = bias[(t + 8) * 16 + n15];
    const float bo = bias[(t + 12) * 16 + n15];
#pragma unroll
    for (int i = 0; i < 8; ++i) { ai[i] = bi; ag[i] = bg; ao[i] = bo; }
    ai = wmma_f16(a, load_b(w, 0, 16, t), ai);
    ag = wmma_f16(a, load_b(w, 0, 16, t + 8), ag);
    ao = wmma_f16(a, load_b(w, 0, 16, t + 12), ao);
    const int n = t * 16 + n15;
#pragma unroll
    for (int v = 0; v < 8; ++v) {
      const float h = sig_(ao[v]) * tanh_(sig_(ai[v]) * tanh_(ag[v]));
      if (!ISFREQ) {
        B[(v + hf * 8) * STRIDE + outCol + n] = (_Float16)h;
      } else {
        const float s = h > LAM ? h - LAM : (h < -LAM ? h + LAM : 0.f);
        const int col = (n <= 32) ? n : n + 1;  // imag[q]=yf[32+q] -> v[33+q]
        B[(v + hf * 8) * STRIDE + outCol + col] = (_Float16)s;
      }
    }
  }
}

// fused 128->12 affine, f32 store to global
__device__ __forceinline__ void fuse_stage(const _Float16* B, int acol,
                                           const _Float16* w, const float* bias,
                                           float* out, int s0) {
  const int lane = threadIdx.x & 31;
  const int n15 = lane & 15, hf = lane >> 4;
  v8f acc;
  const float bv = bias[n15];
#pragma unroll
  for (int i = 0; i < 8; ++i) acc[i] = bv;
#pragma unroll
  for (int kt = 0; kt < 4; ++kt)
    acc = wmma_f16(load_a(B, acol, kt), load_b(w, kt, 1, 0), acc);
  if (n15 < 12) {
#pragma unroll
    for (int v = 0; v < 8; ++v)
      out[(size_t)(s0 + v + hf * 8) * 12 + n15] = acc[v];
  }
}

// ---------------- main kernel: 2 waves/block, 16 samples/wave ----------------
__global__ __launch_bounds__(64) void forecast_kernel(const float* __restrict__ x,
                                                      const _Float16* __restrict__ wp,
                                                      const float* __restrict__ bias,
                                                      float* __restrict__ out) {
  __shared__ _Float16 smem[2 * 16 * STRIDE];  // 48 KB
  const int wave = threadIdx.x >> 5, lane = threadIdx.x & 31;
  _Float16* B = smem + wave * 16 * STRIDE;
  const int s0 = (blockIdx.x * 2 + wave) * 16;

  // warm L2 with packed weights
  __builtin_prefetch(wp + (threadIdx.x << 6), 0, 1);

  // 1. load X (16 samples x 384 feats, k = t*32+c) as f16 into LDS cols 0..383
  //    vectorized: float4 global loads -> v4h (b64) LDS stores
  for (int idx = lane; idx < 16 * 96; idx += 32) {
    const int m = idx / 96, q = idx % 96;
    const float4 f4 = ((const float4*)(x + (size_t)(s0 + m) * 384))[q];
    v4h h;
    h[0] = (_Float16)f4.x; h[1] = (_Float16)f4.y;
    h[2] = (_Float16)f4.z; h[3] = (_Float16)f4.w;
    *(v4h*)&B[m * STRIDE + q * 4] = h;
  }
  wave_lds_sync();

  // F1: (rfft folded into conv1) K=384 -> N=112, pool 14->7 => p1_f at 672 (padK 64)
  gemm_stage<12, 7>(B, 0, wp + W_F1, bias + B_F1, B, 384); wave_lds_sync();
  pool_pass(B, 384, 8, 14, 7, 672, 64); wave_lds_sync();
  // T1: K=384 -> N=96, pool 12->6 => p1_t at 0 (X dead after this GEMM)
  gemm_stage<12, 6>(B, 0, wp + W_T1, bias + B_T1, B, 384); wave_lds_sync();
  pool_pass(B, 384, 8, 12, 6, 0, 64); wave_lds_sync();
  // T2: K=64 -> N=96, pool 6->3 => p2_t at 64
  gemm_stage<2, 6>(B, 0, wp + W_T2, bias + B_T2, B, 384); wave_lds_sync();
  pool_pass(B, 384, 16, 6, 3, 64, 64); wave_lds_sync();
  // T3: K=64 -> N=96, pool 3->1 => p3_t (32) at 128
  gemm_stage<2, 6>(B, 64, wp + W_T3, bias + B_T3, B, 384); wave_lds_sync();
  pool_pass(B, 384, 32, 3, 1, 128, 32); wave_lds_sync();
  // LSTM(T): tf (64) -> cols 512..575
  lstm_stage<false>(B, 128, wp + W_LSTM, bias + B_LSTM, B, 512); wave_lds_sync();
  // F2: K=64 -> N=112, pool 7->3 (drop last) => p2_f at 160
  gemm_stage<2, 7>(B, 672, wp + W_F2, bias + B_F2, B, 384); wave_lds_sync();
  pool_pass(B, 384, 16, 7, 3, 160, 64); wave_lds_sync();
  // F3 (same weights as T3): => p3_f (32) at 224
  gemm_stage<2, 6>(B, 160, wp + W_T3, bias + B_T3, B, 384); wave_lds_sync();
  pool_pass(B, 384, 32, 3, 1, 224, 32); wave_lds_sync();
  // LSTM(F) + softshrink spectrum v[66] at cols 320..385
  lstm_stage<true>(B, 224, wp + W_LSTM, bias + B_LSTM, B, 320);
  // zero v pads: cols 320+{33,65} and 320+[66..95] (K padded to 96)
  for (int idx = lane; idx < 16 * 30; idx += 32) {
    const int m = idx / 30, c = idx % 30;
    B[m * STRIDE + 320 + 66 + c] = (_Float16)0.f;
  }
  if (lane < 16) {
    B[lane * STRIDE + 320 + 33] = (_Float16)0.f;
    B[lane * STRIDE + 320 + 65] = (_Float16)0.f;
  }
  wave_lds_sync();
  // irfft: K=96 -> N=64, ff at 576..639 (z = [tf|ff] contiguous at 512)
  gemm_stage<3, 4>(B, 320, wp + W_MIR, bias + B_ZERO, B, 576); wave_lds_sync();
  // fused 128->12
  fuse_stage(B, 512, wp + W_FUSE, bias + B_FZ, out, s0);
}

// ---------------- weight folding / packing ----------------
__device__ float wval(int wid, int k, int n,
                      const float* c1w, const float* c2w, const float* c3w,
                      const float* wih, const float* f1w, const float* f2w,
                      const float* f3w) {
  switch (wid) {
    case 0: {  // Wt1: k=t*32+c, n=o*12+u ; w1[o,c,l-u+5]
      const int t = k >> 5, c = k & 31, o = n / 12, u = n % 12;
      const int kk = t - u + 5;
      return (kk >= 0 && kk < 12) ? c1w[(o * 32 + c) * 12 + kk] : 0.f;
    }
    case 1: {  // Wt2: k=c*6+l (k<48), n=o*6+u ; w2[o,c,l-u+2]
      if (k >= 48) return 0.f;
      const int c = k / 6, l = k % 6, o = n / 6, u = n % 6;
      const int kk = l - u + 2;
      return (kk >= 0 && kk < 6) ? c2w[(o * 8 + c) * 6 + kk] : 0.f;
    }
    case 2: {  // Wt3: k=c*3+l (k<48), n=o*3+u ; w3[o,c,l-u+1]
      if (k >= 48) return 0.f;
      const int c = k / 3, l = k % 3, o = n / 3, u = n % 3;
      const int kk = l - u + 1;
      return (kk >= 0 && kk < 3) ? c3w[(o * 16 + c) * 3 + kk] : 0.f;
    }
    case 3:  // Wlstm: gates = x @ wih.T
      return wih[n * 32 + k];
    case 4: {  // Wf1 = rfft-expand composed with conv1: k=t*32+c, n=o*14+u
      const int t = k >> 5, c = k & 31, o = n / 14, u = n % 14;
      float s = 0.f;
      for (int l = 0; l < 14; ++l) {
        const int kk = l - u + 5;
        if (kk < 0 || kk >= 12) continue;
        const int q = l >> 1;
        const float ang = PI2 * (float)(q * t) / 12.f;
        const float d = (l & 1) ? -__sinf(ang) : __cosf(ang);
        s += d * c1w[(o * 32 + c) * 12 + kk];
      }
      return s * 0.28867513459f;  // 1/sqrt(12) (ortho rfft)
    }
    case 5: {  // Wf2: k=c*7+l (k<56), n=o*7+u
      if (k >= 56) return 0.f;
      const int c = k / 7, l = k % 7, o = n / 7, u = n % 7;
      const int kk = l - u + 2;
      return (kk >= 0 && kk < 6) ? c2w[(o * 8 + c) * 6 + kk] : 0.f;
    }
    case 6: {  // Mir (ortho irfft, 66->64): rows 0..32 real, 33..65 imag
      if (k >= 66) return 0.f;
      if (k < 33) {
        const float wk = (k == 0 || k == 32) ? 1.f : 2.f;
        return 0.125f * wk * __cosf(PI2 * (float)(k * n) / 64.f);
      }
      const int q = k - 33;
      const float wk = (q == 0 || q == 32) ? 1.f : 2.f;
      return -0.125f * wk * __sinf(PI2 * (float)(q * n) / 64.f);
    }
    case 7: {  // Wfuse = f3 @ f2 @ f1 : (12 x 128), n<12 valid
      if (n >= 12) return 0.f;
      float s = 0.f;
      for (int a = 0; a < 32; ++a) {
        float t2 = 0.f;
        for (int b = 0; b < 64; ++b) t2 += f2w[a * 64 + b] * f1w[b * 128 + k];
        s += f3w[n * 32 + a] * t2;
      }
      return s;
    }
  }
  return 0.f;
}

__global__ void pack_weights(const float* c1w, const float* c2w, const float* c3w,
                             const float* wih, const float* f1w, const float* f2w,
                             const float* f3w, _Float16* wp) {
  const int Ks[8] = {384, 64, 64, 32, 384, 64, 96, 128};
  const int Ns[8] = {96, 96, 96, 256, 112, 112, 64, 16};
  int offs[9];
  offs[0] = 0;
  for (int i = 0; i < 8; ++i) offs[i + 1] = offs[i] + Ks[i] * Ns[i];
  for (int e = blockIdx.x * blockDim.x + threadIdx.x; e < W_TOTAL;
       e += gridDim.x * blockDim.x) {
    int wid = 0;
    while (e >= offs[wid + 1]) ++wid;
    const int local = e - offs[wid];
    const int idx16 = local & 15;          // 2*v + e2 within lane's v16h
    const int lane = (local >> 4) & 31;
    const int frag = local >> 9;
    const int NT = Ns[wid] >> 4;
    const int nt = frag % NT, kt = frag / NT;
    const int v = idx16 >> 1, e2 = idx16 & 1;
    // B-fragment layout: n = nt*16 + lane&15; k = kt*32 + (v&4)*4 + (lane>>4)*8 + (v&3)*2 + e2
    const int kk = kt * 32 + ((v & 4) << 2) + (lane >> 4) * 8 + ((v & 3) << 1) + e2;
    const int nn = nt * 16 + (lane & 15);
    wp[e] = (_Float16)wval(wid, kk, nn, c1w, c2w, c3w, wih, f1w, f2w, f3w);
  }
}

__global__ void pack_bias(const float* c1b, const float* c2b, const float* c3b,
                          const float* bih, const float* bhh, const float* f1w,
                          const float* f1b, const float* f2w, const float* f2b,
                          const float* f3w, const float* f3b, float* bias) {
  const int t = blockIdx.x * blockDim.x + threadIdx.x;
  if (t < 96) bias[t] = c1b[t / 12];
  else if (t < 192) bias[t] = c2b[(t - 96) / 6];
  else if (t < 288) bias[t] = c3b[(t - 192) / 3];
  else if (t < 544) { const int j = t - 288; bias[t] = bih[j] + bhh[j]; }
  else if (t < 656) bias[t] = c1b[(t - 544) / 14];
  else if (t < 768) bias[t] = c2b[(t - 656) / 7];
  else if (t < 784) {
    const int r = t - 768;
    float v = 0.f;
    if (r < 12) {  // f3 @ (f2 @ b1 + b2) + b3
      v = f3b[r];
      for (int a = 0; a < 32; ++a) {
        float inner = f2b[a];
        for (int b = 0; b < 64; ++b) inner += f2w[a * 64 + b] * f1b[b];
        v += f3w[r * 32 + a] * inner;
      }
    }
    bias[t] = v;
  } else if (t < B_TOTAL) {
    bias[t] = 0.f;  // zero-bias block for irfft
  }
}

extern "C" void kernel_launch(void* const* d_in, const int* in_sizes, int n_in,
                              void* d_out, int out_size, void* d_ws, size_t ws_size,
                              hipStream_t stream) {
  const float* x   = (const float*)d_in[0];
  // d_in[1] = x_dec (unused by the reference)
  const float* c1w = (const float*)d_in[2];
  const float* c1b = (const float*)d_in[3];
  const float* c2w = (const float*)d_in[4];
  const float* c2b = (const float*)d_in[5];
  const float* c3w = (const float*)d_in[6];
  const float* c3b = (const float*)d_in[7];
  const float* wih = (const float*)d_in[8];
  // d_in[9] = lstm_whh: dead (single-step scan with h0=0)
  const float* bih = (const float*)d_in[10];
  const float* bhh = (const float*)d_in[11];
  const float* f1w = (const float*)d_in[12];
  const float* f1b = (const float*)d_in[13];
  const float* f2w = (const float*)d_in[14];
  const float* f2b = (const float*)d_in[15];
  const float* f3w = (const float*)d_in[16];
  const float* f3b = (const float*)d_in[17];

  _Float16* wp = (_Float16*)d_ws;
  float* bias  = (float*)((char*)d_ws + (size_t)W_TOTAL * 2);

  pack_weights<<<128, 256, 0, stream>>>(c1w, c2w, c3w, wih, f1w, f2w, f3w, wp);
  pack_bias<<<4, 256, 0, stream>>>(c1b, c2b, c3b, bih, bhh, f1w, f1b, f2w, f2b,
                                   f3w, f3b, bias);
  // 65536 samples / (16 per wave * 2 waves) = 2048 blocks
  forecast_kernel<<<2048, 64, 0, stream>>>(x, wp, bias, (float*)d_out);
}
